// Attention_87076166959630
// MI455X (gfx1250) — matrix operand order repaired and checked by reading, not compiled, not run on hardware
//
#include <hip/hip_runtime.h>
#include <hip/hip_bf16.h>
#include <math.h>

// Problem constants (match reference)
#define NB    2
#define NSEQ  2048
#define DIMX  256
#define NHEAD 8
#define DHEAD 64
#define INNER 512
#define ROWS  (NB * NSEQ)        // 4096
#define QK_SCALE 0.125f          // 64^-0.5
#define NEG_BIG (-3.0e38f)       // ~ -finfo(f32).max like the reference

typedef __attribute__((ext_vector_type(16))) _Float16 v16h;
typedef __attribute__((ext_vector_type(8)))  _Float16 v8h;
typedef __attribute__((ext_vector_type(8)))  float    v8f;
typedef __attribute__((ext_vector_type(4)))  int      v4i;

// gfx1250 async global->LDS copy (ASYNCcnt-tracked), with safe fallback.
// Diagnostic-probed signature: (v4i addrspace(1)*, v4i addrspace(3)*, imm, imm)
#if defined(__gfx1250__) && __has_builtin(__builtin_amdgcn_global_load_async_to_lds_b128) && __has_builtin(__builtin_amdgcn_s_wait_asynccnt)
#define USE_ASYNC_LDS 1
typedef __attribute__((address_space(1))) v4i gas_v4i;
typedef __attribute__((address_space(3))) v4i las_v4i;
#else
#define USE_ASYNC_LDS 0
#endif

__device__ __forceinline__ v8f wmma_f16(v16h a, v16h b, v8f c) {
  // D = A(16x32) * B(32x16) + C(16x16 f32)
  return __builtin_amdgcn_wmma_f32_16x16x32_f16(false, a, false, b, (short)0, c,
                                                false, false);
}

// A-fragment (16x32 f16). Lane l: m = l&15; k = base0 + (e&7) + 16*(e>>3),
// base0 = (l>>4)*8.  Two 16B contiguous LDS loads per lane.
__device__ __forceinline__ v16h frag_a(const _Float16* base, int rstride, int lane) {
  const int m  = lane & 15;
  const int b0 = (lane >> 4) << 3;
  const _Float16* p = base + m * rstride + b0;
  v8h lo = *(const v8h*)p;
  v8h hi = *(const v8h*)(p + 16);
  v16h r;
#pragma unroll
  for (int i = 0; i < 8; ++i) { r[i] = lo[i]; r[i + 8] = hi[i]; }
  return r;
}

// B-fragment (32x16 f16) from k-contiguous storage: row index = n = lane&15,
// elements e -> k = (lane>>4)*8 + (e&7) + 16*(e>>3)  (two 16B loads per lane).
__device__ __forceinline__ v16h frag_b(const _Float16* base, int rstride, int lane) {
  const int n  = lane & 15;
  const int kb = (lane >> 4) << 3;
  const _Float16* p = base + n * rstride + kb;
  v8h lo = *(const v8h*)p;
  v8h hi = *(const v8h*)(p + 16);
  v16h r;
#pragma unroll
  for (int i = 0; i < 8; ++i) { r[i] = lo[i]; r[i + 8] = hi[i]; }
  return r;
}

// Reductions across the 16 lanes of a half-wave (xor masks < 16 stay in-half).
__device__ __forceinline__ float rmax16(float v) {
#pragma unroll
  for (int m = 8; m >= 1; m >>= 1) v = fmaxf(v, __shfl_xor(v, m, 32));
  return v;
}
__device__ __forceinline__ float rsum16(float v) {
#pragma unroll
  for (int m = 8; m >= 1; m >>= 1) v += __shfl_xor(v, m, 32);
  return v;
}

// Pack 16 f32 (as 4 float4) into two v8h halves.
__device__ __forceinline__ void pack16(const float4* f, v8h& lo, v8h& hi) {
#pragma unroll
  for (int i = 0; i < 2; ++i) {
    lo[i * 4 + 0] = (_Float16)f[i].x;     lo[i * 4 + 1] = (_Float16)f[i].y;
    lo[i * 4 + 2] = (_Float16)f[i].z;     lo[i * 4 + 3] = (_Float16)f[i].w;
    hi[i * 4 + 0] = (_Float16)f[i + 2].x; hi[i * 4 + 1] = (_Float16)f[i + 2].y;
    hi[i * 4 + 2] = (_Float16)f[i + 2].z; hi[i * 4 + 3] = (_Float16)f[i + 2].w;
  }
}

// ---------------------------------------------------------------------------
// Kernel 1: fused projections.  [q|k|v|g] = x @ [Wq | Wkv | Wg]  (M=4096,K=256)
// ---------------------------------------------------------------------------
__global__ void __launch_bounds__(128) proj_kernel(
    const float* __restrict__ x,
    const float* __restrict__ Wq, const float* __restrict__ Wkv,
    const float* __restrict__ Wg, const float* __restrict__ bg,
    _Float16* __restrict__ qh, _Float16* __restrict__ kh,
    _Float16* __restrict__ vh, _Float16* __restrict__ gh)
{
  __shared__ _Float16 As [64 * 40];   // x tile [64 rows][32 k], pad->40
  __shared__ _Float16 BsT[64 * 40];   // W tile transposed [64 n][32 k], pad->40
  const int tid  = threadIdx.x;
  const int lane = tid & 31;
  const int w    = tid >> 5;
  const int n0   = blockIdx.x * 64;   // 0..2047
  const int m0   = blockIdx.y * 64;   // 0..4095

  // Uniform (SGPR-only) weight-segment selection: a 64-wide n-tile cannot
  // straddle a 512-aligned segment boundary, so branch on n0 alone.
  const float* wseg;   // base of selected weight matrix
  size_t       wld;    // its leading dimension (floats)
  int          ncb;    // column of this tile within the selected matrix
  if (n0 < 512)       { wseg = Wq;  wld = 512;  ncb = n0; }
  else if (n0 < 1536) { wseg = Wkv; wld = 1024; ncb = n0 - 512; }
  else                { wseg = Wg;  wld = 512;  ncb = n0 - 1536; }

  v8f acc[4] = {};

  for (int k0 = 0; k0 < DIMX; k0 += 32) {
    __syncthreads();
    { // stage A: 64x32 f32 -> f16, coalesced float4 reads, packed b128 stores
      const int row = tid >> 1;
      const int kb  = (tid & 1) << 4;
      const float* xp = x + (size_t)(m0 + row) * DIMX + k0 + kb;
      float4 f[4];
#pragma unroll
      for (int i = 0; i < 4; ++i) f[i] = *(const float4*)(xp + i * 4);
      v8h lo, hi;
      pack16(f, lo, hi);
      *(v8h*)&As[row * 40 + kb]     = lo;
      *(v8h*)&As[row * 40 + kb + 8] = hi;
    }
    { // stage B transposed: read W rows coalesced, scatter into [n][k]
      const int kr   = tid >> 2;            // 0..31
      const int cblk = (tid & 3) << 4;      // 0,16,32,48
      const float* wrow = wseg + (size_t)(k0 + kr) * wld + ncb + cblk;
#pragma unroll
      for (int i = 0; i < 4; ++i) {
        float4 f = *(const float4*)(wrow + i * 4);
        BsT[(cblk + i * 4 + 0) * 40 + kr] = (_Float16)f.x;
        BsT[(cblk + i * 4 + 1) * 40 + kr] = (_Float16)f.y;
        BsT[(cblk + i * 4 + 2) * 40 + kr] = (_Float16)f.z;
        BsT[(cblk + i * 4 + 3) * 40 + kr] = (_Float16)f.w;
      }
    }
    __syncthreads();

    v16h a = frag_a(As + w * 16 * 40, 40, lane);
#pragma unroll
    for (int t = 0; t < 4; ++t) {
      v16h bf = frag_b(BsT + t * 16 * 40, 40, lane);
      acc[t] = wmma_f16(a, bf, acc[t]);
    }
  }

  // epilogue: route to q/k/v/g (seg is uniform: derived from n0 only)
  const int seg   = n0 >> 9;
  const int nbase = n0 & 511;
  const int hfw   = lane >> 4;
  const int nlan  = lane & 15;
#pragma unroll
  for (int t = 0; t < 4; ++t) {
#pragma unroll
    for (int r = 0; r < 8; ++r) {
      const int m  = m0 + w * 16 + r + 8 * hfw;
      const int nc = nbase + t * 16 + nlan;
      const float v = acc[t][r];
      const size_t idx = (size_t)m * INNER + nc;
      if (seg == 0)      qh[idx] = (_Float16)(v * QK_SCALE);
      else if (seg == 1) kh[idx] = (_Float16)v;
      else if (seg == 2) vh[idx] = (_Float16)v;
      else               gh[idx] = (_Float16)(v + bg[nc]);
    }
  }
}

// ---------------------------------------------------------------------------
// Kernel 2: flash attention per (b,h).  Block = 4 waves, each owns a 16-row
// i-tile.  j loop in chunks of 32 with LDS-shared K (async copy) / V (manual
// transposed staging).  Per chunk per wave: 4 QK^T WMMAs + 4 PV WMMAs.
// ---------------------------------------------------------------------------
__global__ void __launch_bounds__(128) attn_kernel(
    const _Float16* __restrict__ qh, const _Float16* __restrict__ kh,
    const _Float16* __restrict__ vh, const float* __restrict__ bias,
    const unsigned char* __restrict__ mask, _Float16* __restrict__ oh)
{
  __shared__ _Float16 Ks [32 * 72];     // K chunk [32 j][64 d], pad->72
  __shared__ _Float16 VsT[64 * 40];     // V chunk transposed [64 d][32 j], pad->40
  __shared__ _Float16 Ps[4][16 * 40];   // per-wave P tile [16 m][32 j], pad->40

  const int tid  = threadIdx.x;
  const int lane = tid & 31;
  const int w    = tid >> 5;
  const int b    = (int)blockIdx.x >> 3;
  const int h    = (int)blockIdx.x & 7;
  const int i0   = (int)blockIdx.y * 64 + w * 16;
  const int hfw  = lane >> 4;
  const int nlan = lane & 15;

  // Q fragments (d=0..31 and d=32..63); q already scaled by 1/sqrt(d)
  v16h qa0, qa1;
  {
    const int b0 = hfw << 3;
    const _Float16* p =
        qh + (size_t)(b * NSEQ + i0 + nlan) * INNER + h * DHEAD + b0;
    v8h l0 = *(const v8h*)p;
    v8h h0 = *(const v8h*)(p + 16);
    v8h l1 = *(const v8h*)(p + 32);
    v8h h1 = *(const v8h*)(p + 48);
#pragma unroll
    for (int i = 0; i < 8; ++i) {
      qa0[i] = l0[i]; qa0[i + 8] = h0[i];
      qa1[i] = l1[i]; qa1[i + 8] = h1[i];
    }
  }

  bool mi[8];
#pragma unroll
  for (int r = 0; r < 8; ++r)
    mi[r] = mask[b * NSEQ + i0 + r + 8 * hfw] != 0;

  float mrow[8], lrow[8];
#pragma unroll
  for (int r = 0; r < 8; ++r) { mrow[r] = NEG_BIG; lrow[r] = 0.f; }
  v8f acc[4] = {};

  const float* bias_bh = bias + ((size_t)(b * NHEAD + h) * NSEQ + i0) * NSEQ;

  for (int j0 = 0; j0 < NSEQ; j0 += 32) {
    __syncthreads();
    { // cooperative stage: K row-major (async DMA to LDS), V transposed
      const int jr   = tid >> 2;           // 0..31
      const int dblk = (tid & 3) << 4;     // 0,16,32,48
      const size_t g = (size_t)(b * NSEQ + j0 + jr) * INNER + h * DHEAD + dblk;
#if USE_ASYNC_LDS
      __builtin_amdgcn_global_load_async_to_lds_b128(
          (gas_v4i*)(kh + g), (las_v4i*)&Ks[jr * 72 + dblk], 0, 0);
      __builtin_amdgcn_global_load_async_to_lds_b128(
          (gas_v4i*)(kh + g + 8), (las_v4i*)&Ks[jr * 72 + dblk + 8], 0, 0);
#else
      v8h k0v = *(const v8h*)(kh + g);
      v8h k1v = *(const v8h*)(kh + g + 8);
      *(v8h*)&Ks[jr * 72 + dblk]     = k0v;
      *(v8h*)&Ks[jr * 72 + dblk + 8] = k1v;
#endif
      v8h v0v = *(const v8h*)(vh + g);
      v8h v1v = *(const v8h*)(vh + g + 8);
#pragma unroll
      for (int i = 0; i < 8; ++i) {
        VsT[(dblk + i)     * 40 + jr] = v0v[i];
        VsT[(dblk + 8 + i) * 40 + jr] = v1v[i];
      }
#if USE_ASYNC_LDS
      __builtin_amdgcn_s_wait_asynccnt(0);
#endif
    }
    __syncthreads();

    // S = (q/sqrt(d)) . k^T  : two 16x16 tiles, each = 2 chained WMMAs (d=64)
    v8f s[2];
#pragma unroll
    for (int t = 0; t < 2; ++t) {
      v16h kb0 = frag_b(Ks + t * 16 * 72,      72, lane);  // d 0..31
      v16h kb1 = frag_b(Ks + t * 16 * 72 + 32, 72, lane);  // d 32..63
      v8f z = {};
      z = wmma_f16(qa0, kb0, z);
      z = wmma_f16(qa1, kb1, z);
      s[t] = z;
    }

    // + bias (non-temporal 268MB stream), apply pair mask
    bool mj[2];
#pragma unroll
    for (int t = 0; t < 2; ++t)
      mj[t] = mask[b * NSEQ + j0 + t * 16 + nlan] != 0;
#pragma unroll
    for (int t = 0; t < 2; ++t) {
      const float* bp = bias_bh + (size_t)(8 * hfw) * NSEQ + j0 + t * 16 + nlan;
#pragma unroll
      for (int r = 0; r < 8; ++r) {
        float bv = __builtin_nontemporal_load(bp + (size_t)r * NSEQ);
        float sv = s[t][r] + bv;
        if (!(mi[r] && mj[t])) sv = NEG_BIG;
        s[t][r] = sv;
      }
    }

    // online softmax (row stats replicated across the 16 lanes of a half-wave)
#pragma unroll
    for (int r = 0; r < 8; ++r) {
      float mx = rmax16(fmaxf(s[0][r], s[1][r]));
      float nm = fmaxf(mrow[r], mx);
      float al = __expf(mrow[r] - nm);
      mrow[r] = nm;
      float p0 = __expf(s[0][r] - nm);
      float p1 = __expf(s[1][r] - nm);
      s[0][r] = p0; s[1][r] = p1;
      lrow[r] = lrow[r] * al + rsum16(p0 + p1);
#pragma unroll
      for (int t2 = 0; t2 < 4; ++t2) acc[t2][r] *= al;
    }

    // P (C-layout f32) -> per-wave LDS f16 tile -> A-fragment; then O += P@V
    _Float16* ps = &Ps[w][0];
#pragma unroll
    for (int t = 0; t < 2; ++t)
#pragma unroll
      for (int r = 0; r < 8; ++r)
        ps[(r + 8 * hfw) * 40 + t * 16 + nlan] = (_Float16)s[t][r];
    // same-wave LDS RAW: DS ops are in-order; compiler inserts s_wait_dscnt

    v16h pa = frag_a(ps, 40, lane);
#pragma unroll
    for (int t2 = 0; t2 < 4; ++t2) {
      v16h vb = frag_b(VsT + t2 * 16 * 40, 40, lane);
      acc[t2] = wmma_f16(pa, vb, acc[t2]);
    }
  }

  // normalize and write attention output [b, n, h*64+d] as f16
#pragma unroll
  for (int r = 0; r < 8; ++r) {
    const float inv = 1.0f / lrow[r];
    const int m = i0 + r + 8 * hfw;
    _Float16* op = oh + (size_t)(b * NSEQ + m) * INNER + h * DHEAD + nlan;
#pragma unroll
    for (int t2 = 0; t2 < 4; ++t2)
      op[t2 * 16] = (_Float16)(acc[t2][r] * inv);
  }
}

// ---------------------------------------------------------------------------
// Kernel 3: out = (attn_out .* gates) @ Wout + bout   (M=4096,K=512,N=256)
// ---------------------------------------------------------------------------
__global__ void __launch_bounds__(128) out_kernel(
    const _Float16* __restrict__ oh, const _Float16* __restrict__ gh,
    const float* __restrict__ Wout, const float* __restrict__ bout,
    float* __restrict__ out)
{
  __shared__ _Float16 As [64 * 40];
  __shared__ _Float16 BsT[64 * 40];
  const int tid  = threadIdx.x;
  const int lane = tid & 31;
  const int w    = tid >> 5;
  const int n0   = blockIdx.x * 64;   // 0..255
  const int m0   = blockIdx.y * 64;   // 0..4095
  v8f acc[4] = {};

  for (int k0 = 0; k0 < INNER; k0 += 32) {
    __syncthreads();
    { // stage A = attn_out * gates (gating fused here), packed b128 stores
      const int row = tid >> 1;
      const int kb  = (tid & 1) << 4;
      const size_t g = (size_t)(m0 + row) * INNER + k0 + kb;
      v8h o0 = *(const v8h*)(oh + g), o1 = *(const v8h*)(oh + g + 8);
      v8h g0 = *(const v8h*)(gh + g), g1 = *(const v8h*)(gh + g + 8);
      v8h r0, r1;
#pragma unroll
      for (int i = 0; i < 8; ++i) {
        r0[i] = (_Float16)((float)o0[i] * (float)g0[i]);
        r1[i] = (_Float16)((float)o1[i] * (float)g1[i]);
      }
      *(v8h*)&As[row * 40 + kb]     = r0;
      *(v8h*)&As[row * 40 + kb + 8] = r1;
    }
    { // stage Wout transposed
      const int kr   = tid >> 2;
      const int cblk = (tid & 3) << 4;
      const float* wrow = Wout + (size_t)(k0 + kr) * DIMX + n0 + cblk;
#pragma unroll
      for (int i = 0; i < 4; ++i) {
        float4 f = *(const float4*)(wrow + i * 4);
        BsT[(cblk + i * 4 + 0) * 40 + kr] = (_Float16)f.x;
        BsT[(cblk + i * 4 + 1) * 40 + kr] = (_Float16)f.y;
        BsT[(cblk + i * 4 + 2) * 40 + kr] = (_Float16)f.z;
        BsT[(cblk + i * 4 + 3) * 40 + kr] = (_Float16)f.w;
      }
    }
    __syncthreads();

    v16h a = frag_a(As + w * 16 * 40, 40, lane);
#pragma unroll
    for (int t = 0; t < 4; ++t) {
      v16h bf = frag_b(BsT + t * 16 * 40, 40, lane);
      acc[t] = wmma_f16(a, bf, acc[t]);
    }
  }

  const int hfw  = lane >> 4;
  const int nlan = lane & 15;
#pragma unroll
  for (int t = 0; t < 4; ++t) {
    const int nc = n0 + t * 16 + nlan;
    const float bb = bout[nc];
#pragma unroll
    for (int r = 0; r < 8; ++r) {
      const int m = m0 + w * 16 + r + 8 * hfw;
      out[(size_t)m * DIMX + nc] = acc[t][r] + bb;
    }
  }
}

// ---------------------------------------------------------------------------
extern "C" void kernel_launch(void* const* d_in, const int* in_sizes, int n_in,
                              void* d_out, int out_size, void* d_ws, size_t ws_size,
                              hipStream_t stream) {
  (void)in_sizes; (void)n_in; (void)out_size; (void)ws_size;
  const float*         x    = (const float*)d_in[0];
  const unsigned char* mask = (const unsigned char*)d_in[1];  // jnp bool = 1 byte
  const float*         bias = (const float*)d_in[2];
  const float*         Wq   = (const float*)d_in[3];
  const float*         Wkv  = (const float*)d_in[4];
  const float*         Wout = (const float*)d_in[5];
  const float*         bout = (const float*)d_in[6];
  const float*         Wg   = (const float*)d_in[7];
  const float*         bg   = (const float*)d_in[8];
  float* out = (float*)d_out;

  // workspace: 5 f16 arrays of [4096][512] = 20 MB total
  const size_t stride = (size_t)ROWS * INNER;
  _Float16* qh = (_Float16*)d_ws;
  _Float16* kh = qh + stride;
  _Float16* vh = kh + stride;
  _Float16* gh = vh + stride;
  _Float16* oh = gh + stride;

  proj_kernel<<<dim3(2048 / 64, ROWS / 64), 128, 0, stream>>>(
      x, Wq, Wkv, Wg, bg, qh, kh, vh, gh);
  attn_kernel<<<dim3(NB * NHEAD, NSEQ / 64), 128, 0, stream>>>(
      qh, kh, vh, bias, mask, oh);
  out_kernel<<<dim3(DIMX / 64, ROWS / 64), 128, 0, stream>>>(
      oh, gh, Wout, bout, out);
}
